// YOLOLayer_70411693851327
// MI455X (gfx1250) — compile-verified, matrix operand order checked by reading, not run on hardware
//
#include <hip/hip_runtime.h>
#include <hip/hip_bf16.h>

// ---------------- problem constants ----------------
#define DIM_S 32
#define DIM_A 3
#define DIM_H 52
#define DIM_W 52
#define N_ATTR 85          // 5 + 80
#define N_CLS 80
#define PLANE (DIM_H * DIM_W)          // 2704
#define NPOS  (DIM_S * DIM_A * DIM_H * DIM_W)   // 259584
#define PRED_N (NPOS * 6)              // 1557504
#define K_NMS 1024
#define OBJ_THR 0.9f
#define NMS_THR 0.5f
#define CAP 16384                      // candidate capacity (mean ~3.6K, +200 sigma)

typedef _Float16 v16h __attribute__((ext_vector_type(16)));
typedef float    v8f  __attribute__((ext_vector_type(8)));
typedef unsigned int v4u __attribute__((ext_vector_type(4)));
typedef int      v8i  __attribute__((ext_vector_type(8)));
typedef int      v4i  __attribute__((ext_vector_type(4)));

__device__ __forceinline__ float sigm(float v) { return 1.0f / (1.0f + expf(-v)); }

// ---------------- TDM 2D tile load: global -> LDS ----------------
// tile: tile_d1 lines x tile_d0 f32 elements, line stride stride0 (elements).
__device__ __forceinline__ void tdm_load_2d_f32(const void* gptr, unsigned lds_off,
                                                unsigned tile_d0, unsigned tile_d1,
                                                unsigned tensor_d0, unsigned tensor_d1,
                                                unsigned long long stride0) {
  unsigned long long ga = (unsigned long long)gptr;
  // D# group0 (ISA 8.3): count=1, lds_addr, global_addr[56:0], type=2
  v4u g0 = { 1u,
             lds_off,
             (unsigned)(ga & 0xffffffffull),
             (unsigned)((ga >> 32) & 0x01ffffffull) | 0x80000000u };
  // D# group1 (ISA 8.4): wg_mask=0, data_size=2 (4B), dims/strides in elements
  v8i g1 = { (int)(2u << 16),
             (int)((tensor_d0 & 0xffffu) << 16),
             (int)((tensor_d0 >> 16) | ((tensor_d1 & 0xffffu) << 16)),
             (int)((tensor_d1 >> 16) | ((tile_d0 & 0xffffu) << 16)),
             (int)(tile_d1 & 0xffffu),                       // tile_dim1; tile_dim2=0
             (int)(stride0 & 0xffffffffull),                 // dim0 stride lo32
             (int)((stride0 >> 32) & 0xffffull),             // dim0 stride hi16; dim1 stride=0
             0 };
  v4i z4 = { 0, 0, 0, 0 };
#if defined(__clang_major__) && (__clang_major__ >= 23)
  v8i z8 = { 0, 0, 0, 0, 0, 0, 0, 0 };
  __builtin_amdgcn_tensor_load_to_lds(g0, g1, z4, z4, z8, 0);
#else
  __builtin_amdgcn_tensor_load_to_lds(g0, g1, z4, z4, 0);
#endif
}

// ---------------- kernel 0: init workspace ----------------
__global__ void k_init(unsigned* counter, int* slot_sel, float* selscore) {
  int t = threadIdx.x;
  if (t == 0) *counter = 0u;
  if (t < K_NMS) { slot_sel[t] = 0; selscore[t] = 0.0f; }
}

// ---------------- kernel 1: decode via TDM staging ----------------
// one block per (s,h): TDM pulls [255 lines x 52 floats] (all anchors, all attrs of one row)
__global__ __launch_bounds__(256) void k_decode(const float* __restrict__ x,
                                                const float* __restrict__ anchors,
                                                float* __restrict__ pred,
                                                unsigned* __restrict__ counter,
                                                int* __restrict__ cand_idx,
                                                float* __restrict__ cand_score) {
  __shared__ float lds[DIM_A * N_ATTR * DIM_W];   // 255*52 floats = 53,040 B
  const int sh = blockIdx.x;
  const int s = sh / DIM_H;
  const int h = sh % DIM_H;
  const float* gsrc = x + (size_t)s * (DIM_A * N_ATTR) * PLANE + (size_t)h * DIM_W;

  if (threadIdx.x < 32) {   // one wave issues the DMA (EXEC ignored by TDM; avoid duplicates)
    unsigned lds_off = (unsigned)(unsigned long long)(uintptr_t)(&lds[0]);
    tdm_load_2d_f32(gsrc, lds_off,
                    /*tile_d0=*/DIM_W, /*tile_d1=*/DIM_A * N_ATTR,
                    /*tensor_d0=*/PLANE, /*tensor_d1=*/DIM_A * N_ATTR,
                    /*stride0=*/PLANE);
    __builtin_amdgcn_s_wait_tensorcnt(0);
  }
  __syncthreads();

  const int t = threadIdx.x;
  if (t < DIM_A * DIM_W) {
    const int a = t / DIM_W;
    const int w = t % DIM_W;
    const float* base = lds + (a * N_ATTR) * DIM_W + w;   // attr stride = DIM_W floats
    const float p0 = base[0 * DIM_W];
    const float p1 = base[1 * DIM_W];
    const float p2 = base[2 * DIM_W];
    const float p3 = base[3 * DIM_W];
    const float p4 = base[4 * DIM_W];
    float best = base[5 * DIM_W];
    int bi = 0;
    #pragma unroll 4
    for (int c = 1; c < N_CLS; ++c) {
      float v = base[(5 + c) * DIM_W];
      if (v > best) { best = v; bi = c; }   // strict > : first-occurrence argmax
    }
    const float scale = 416.0f / (float)DIM_W;   // 8
    const float bx = (sigm(p0) + (float)w) * scale;
    const float by = (sigm(p1) + (float)h) * scale;
    const float bw = anchors[a * 2 + 0] * expf(p2);   // (anchor/8)*exp*8
    const float bh = anchors[a * 2 + 1] * expf(p3);
    const float conf = sigm(p4);
    const size_t fi = ((size_t)(s * DIM_A + a) * DIM_H + h) * DIM_W + w;
    float* o = pred + fi * 6;
    o[0] = bx; o[1] = by; o[2] = bw; o[3] = bh; o[4] = conf; o[5] = (float)bi;
    if (conf > OBJ_THR) {
      unsigned pos = atomicAdd(counter, 1u);
      if (pos < CAP) { cand_idx[pos] = (int)fi; cand_score[pos] = conf; }
    }
  }
}

// ---------------- kernel 2: WMMA rank counting (top-K selection) ----------------
// rank_i = #{ j : s_j > s_i  or (s_j == s_i and idx_j < idx_i) }  -> slot rank_i if < 1024.
// Indicators (0/1, exact in f16) summed with v_wmma_f32_16x16x32_f16 against ones-column B.
// Chunk data is loaded once per wave (coalesced) and spread with shuffles: no divergent
// branches anywhere in the K loop, so EXEC stays all-ones between WMMAs.
__global__ __launch_bounds__(256) void k_rank(const unsigned* __restrict__ counter,
                                              const int* __restrict__ cand_idx,
                                              const float* __restrict__ cand_score,
                                              int* __restrict__ slot_sel,
                                              float* __restrict__ selscore) {
  int C = (int)*counter; if (C > CAP) C = CAP;
  if (C == 0) return;
  const int lane = threadIdx.x & 31;
  const int wid  = threadIdx.x >> 5;
  const int hi   = lane >> 4;      // A-matrix lane half (K offset +8); D: M offset +8
  const int r    = lane & 15;      // A row within tile; D column N
  const int RT   = (C + 15) >> 4;

  v16h bOnes = {};
  if (r == 0) {                    // B 32x16: lanes 0/16 hold column N=0 for K=0..7,16..23 / 8..15,24..31
    #pragma unroll
    for (int e = 0; e < 16; ++e) bOnes[e] = (_Float16)1.0f;
  }

  for (int rt = blockIdx.x * 8 + wid; rt < RT; rt += gridDim.x * 8) {   // wave-uniform
    const int i   = rt * 16 + r;
    const int icl = (i < C) ? i : (C - 1);
    const float si = cand_score[icl];
    const int   ii = cand_idx[icl];
    v8f acc = {};
    for (int j0 = 0; j0 < C; j0 += 32) {
      // one coalesced chunk load per wave; lane l holds entry j0+l
      const int jl  = j0 + lane;
      const int jlc = (jl < C) ? jl : (C - 1);
      const float sjl = cand_score[jlc];
      const int   ijl = cand_idx[jlc];
      v16h aM;
      #pragma unroll
      for (int vv = 0; vv < 8; ++vv) {
        // A 16x32 f16 layout: VGPR vv<4 -> K={2vv,2vv+1}; vv>=4 -> K={16+2(vv-4),..}; lanes16-31 +8
        const int kb = ((vv < 4) ? (2 * vv) : (16 + 2 * (vv - 4))) + hi * 8;
        const float sj0 = __shfl(sjl, kb, 32);
        const int   ij0 = __shfl(ijl, kb, 32);
        const float sj1 = __shfl(sjl, kb + 1, 32);
        const int   ij1 = __shfl(ijl, kb + 1, 32);
        // branchless indicators (bitwise bool ops -> v_cmp/v_cndmask, no EXEC branches)
        const bool v0 = ((j0 + kb)     < C) & ((sj0 > si) | ((sj0 == si) & (ij0 < ii)));
        const bool v1 = ((j0 + kb + 1) < C) & ((sj1 > si) | ((sj1 == si) & (ij1 < ii)));
        aM[2 * vv]     = v0 ? (_Float16)1.0f : (_Float16)0.0f;
        aM[2 * vv + 1] = v1 ? (_Float16)1.0f : (_Float16)0.0f;
      }
      acc = __builtin_amdgcn_wmma_f32_16x16x32_f16(false, aM, false, bOnes,
                                                   (short)0, acc, false, false);
    }
    if (r == 0) {                 // D column 0: lane0 -> rows rt*16+m, lane16 -> rows rt*16+8+m
      #pragma unroll
      for (int m = 0; m < 8; ++m) {
        const int row  = rt * 16 + m + hi * 8;
        const int rank = (int)(acc[m] + 0.5f);
        if (row < C && rank < K_NMS) {
          slot_sel[rank] = cand_idx[row];
          selscore[rank] = cand_score[row];
        }
      }
    }
  }
}

// ---------------- kernel 3: fill remaining slots with score-0 entries (index ascending) ----------------
__global__ void k_zerofill(const float* __restrict__ pred,
                           const unsigned* __restrict__ counter,
                           const int* __restrict__ cand_idx,
                           int* __restrict__ slot_sel,
                           float* __restrict__ selscore) {
  int C = (int)*counter; if (C > CAP) C = CAP;
  const int Cc = (C < K_NMS) ? C : K_NMS;
  if (Cc >= K_NMS) return;
  long long jmax = (long long)K_NMS + C + 1;
  if (jmax > (long long)NPOS) jmax = NPOS;
  for (long long j = (long long)blockIdx.x * blockDim.x + threadIdx.x; j < jmax;
       j += (long long)gridDim.x * blockDim.x) {
    const float conf = pred[j * 6 + 4];
    if (conf > OBJ_THR) continue;            // positive, handled by k_rank
    int pb = 0;                              // #positives with flat index < j
    for (int t = 0; t < C; ++t) pb += (cand_idx[t] < (int)j);
    const long long slot = (long long)Cc + (j - pb);   // zeros sorted by index asc
    if (slot < K_NMS) { slot_sel[slot] = (int)j; selscore[slot] = 0.0f; }
  }
}

// ---------------- kernel 4: greedy NMS (single 1024-thread block) ----------------
__global__ __launch_bounds__(1024) void k_nms(const float* __restrict__ pred,
                                              const int* __restrict__ slot_sel,
                                              const float* __restrict__ selscore,
                                              float* __restrict__ boxes,
                                              float* __restrict__ keep) {
  __shared__ float sX1[K_NMS], sY1[K_NMS], sX2[K_NMS], sY2[K_NMS], sAR[K_NMS], sSC[K_NMS];
  __shared__ int sSUP[K_NMS];
  const int k = threadIdx.x;
  const int idx = slot_sel[k];
  const float* p = pred + (size_t)idx * 6;
  const float bx = p[0], by = p[1], bw = p[2], bh = p[3], cf = p[4], cl = p[5];
  float* bo = boxes + k * 6;
  bo[0] = bx; bo[1] = by; bo[2] = bw; bo[3] = bh; bo[4] = cf; bo[5] = cl;
  const float x1 = bx - bw * 0.5f, y1 = by - bh * 0.5f;
  const float x2 = bx + bw * 0.5f, y2 = by + bh * 0.5f;
  const float myArea = (x2 - x1) * (y2 - y1);
  sX1[k] = x1; sY1[k] = y1; sX2[k] = x2; sY2[k] = y2;
  sAR[k] = myArea; sSC[k] = selscore[k]; sSUP[k] = 0;
  bool mykeep = false;
  __syncthreads();
  for (int i = 0; i < K_NMS; ++i) {
    const bool valid = (sSUP[i] == 0) && (sSC[i] > 0.0f);
    const float xi1 = sX1[i], yi1 = sY1[i], xi2 = sX2[i], yi2 = sY2[i], ai = sAR[i];
    __syncthreads();                        // reads of sSUP[i] precede this round's writes
    if (i == k) mykeep = valid;
    if (valid) {
      float iw = fminf(xi2, x2) - fmaxf(xi1, x1); iw = iw > 0.0f ? iw : 0.0f;
      float ih = fminf(yi2, y2) - fmaxf(yi1, y1); ih = ih > 0.0f ? ih : 0.0f;
      const float inter = iw * ih;
      const float iou = inter / (ai + myArea - inter + 1e-9f);
      if (iou > NMS_THR) sSUP[k] = 1;       // includes self-suppression (iou(i,i)=1), as reference
    }
    __syncthreads();                        // writes visible before next round's reads
  }
  keep[k] = mykeep ? 1.0f : 0.0f;
}

// ---------------- host launcher ----------------
extern "C" void kernel_launch(void* const* d_in, const int* in_sizes, int n_in,
                              void* d_out, int out_size, void* d_ws, size_t ws_size,
                              hipStream_t stream) {
  const float* x = (const float*)d_in[0];
  const float* anchors = (const float*)d_in[1];
  float* out = (float*)d_out;

  float* pred  = out;                       // [S,A,H,W,6]
  float* boxes = out + PRED_N;              // [K,6]
  float* keep  = out + PRED_N + K_NMS * 6;  // [K]

  char* ws = (char*)d_ws;
  unsigned* counter  = (unsigned*)(ws + 0);
  int*   cand_idx    = (int*)  (ws + 64);
  float* cand_score  = (float*)(ws + 64 + (size_t)CAP * 4);
  int*   slot_sel    = (int*)  (ws + 64 + (size_t)CAP * 8);
  float* selscore    = (float*)(ws + 64 + (size_t)CAP * 8 + (size_t)K_NMS * 4);

  k_init<<<1, 1024, 0, stream>>>(counter, slot_sel, selscore);
  k_decode<<<DIM_S * DIM_H, 256, 0, stream>>>(x, anchors, pred, counter, cand_idx, cand_score);
  k_rank<<<256, 256, 0, stream>>>(counter, cand_idx, cand_score, slot_sel, selscore);
  k_zerofill<<<128, 256, 0, stream>>>(pred, counter, cand_idx, slot_sel, selscore);
  k_nms<<<1, 1024, 0, stream>>>(pred, slot_sel, selscore, boxes, keep);
}
